// ReasonerModel_28570122453752
// MI455X (gfx1250) — compile-verified, matrix operand order checked by reading, not run on hardware
//
#include <hip/hip_runtime.h>

// Problem dims (fixed by the reference)
#define Bv   32
#define SQv  80
#define SKVv 1024
#define Dv   1024
#define Hv   16
#define Lv   12
#define HDv  64
#define QTv  (SQv / 16)   // 5 query tiles of 16
#define NKV  (Bv * SKVv)  // 32768

typedef __bf16 bf16;
typedef __bf16 v16bf __attribute__((ext_vector_type(16)));
typedef float  v8f   __attribute__((ext_vector_type(8)));
typedef unsigned int u32x4 __attribute__((ext_vector_type(4)));
typedef int i32x4 __attribute__((ext_vector_type(4)));
typedef int i32x8 __attribute__((ext_vector_type(8)));

#define FLAG_GELU     1
#define FLAG_OUT_BF16 2
#define FLAG_KV_SPLIT 4

static __device__ __forceinline__ v8f wmma_bf16(v16bf a, v16bf b, v8f c) {
  return __builtin_amdgcn_wmma_f32_16x16x32_bf16(false, a, false, b, (short)0, c,
                                                 false, false);
}

static __device__ __forceinline__ float gelu_f(float x) {
  return 0.5f * x * (1.0f + tanhf(0.7978845608028654f * (x + 0.044715f * x * x * x)));
}

// --- Tensor Data Mover: 2D bf16 tile global -> LDS (D# per ISA 08 §8.3/8.4) --
// tile0 = contiguous elements per row, tile1 = rows, stride0 = row stride (elems)
static __device__ __forceinline__ void tdm_load_2d(const bf16* gsrc, unsigned lds_byte,
                                                   unsigned rowlen, unsigned nrows,
                                                   unsigned tile0, unsigned tile1,
                                                   unsigned long long stride0) {
  unsigned long long ga = (unsigned long long)(size_t)gsrc;
  u32x4 g0;
  g0[0] = 1u;                                              // count=1, user mode
  g0[1] = lds_byte;                                        // lds_addr
  g0[2] = (unsigned)ga;                                    // global_addr[31:0]
  g0[3] = (unsigned)((ga >> 32) & 0x01ffffffu) | (2u << 30); // ga[56:32] | type=2
  i32x8 g1;
  g1[0] = 1 << 16;                                         // wg_mask=0, data_size=1 (2B)
  g1[1] = (int)((rowlen & 0xffffu) << 16);                 // tensor_dim0[15:0]
  g1[2] = (int)((rowlen >> 16) | ((nrows & 0xffffu) << 16)); // td0[31:16] | td1[15:0]
  g1[3] = (int)((nrows >> 16) | (tile0 << 16));            // td1[31:16] | tile_dim0
  g1[4] = (int)(tile1 & 0xffffu);                          // tile_dim1 (tile_dim2=0)
  g1[5] = (int)(stride0 & 0xffffffffu);                    // tensor_dim0_stride[31:0]
  g1[6] = (int)((stride0 >> 32) & 0xffffu);                // stride[47:32] | s1[15:0]=0
  g1[7] = 0;
  i32x4 z4 = {};
#if defined(__clang_major__) && (__clang_major__ >= 23)
  i32x8 z8 = {};
  __builtin_amdgcn_tensor_load_to_lds(g0, g1, z4, z4, z8, 0);
#else
  __builtin_amdgcn_tensor_load_to_lds(g0, g1, z4, z4, 0);
#endif
}

// ---------------- fp32 -> bf16 (dense) -------------------------------------
__global__ void k_cvt(const float* __restrict__ src, bf16* __restrict__ dst,
                      long total) {
  long i = (long)blockIdx.x * blockDim.x + threadIdx.x;
  if (i >= total) return;
  dst[i] = (bf16)src[i];
}

// ---- fp32 [L][K][N-slice] -> bf16 transposed [L][N][K] --------------------
__global__ void k_cvtT(const float* __restrict__ src, bf16* __restrict__ dst,
                       long total, int K, int N, int stride, int coff) {
  long i = (long)blockIdx.x * blockDim.x + threadIdx.x;
  if (i >= total) return;
  int  k = (int)(i % K);
  long r = i / K;
  int  n = (int)(r % N);
  int  l = (int)(r / N);
  dst[i] = (bf16)src[((long)l * K + k) * stride + coff + n];
}

// ---------------- h = inputs_embeds + pos_embed (fp32 + bf16 shadow) -------
__global__ void k_add_pos(const float* __restrict__ e, const float* __restrict__ pos,
                          float* __restrict__ h, bf16* __restrict__ hbf, int total) {
  int i = blockIdx.x * blockDim.x + threadIdx.x;
  if (i >= total) return;
  float v = e[i] + pos[i % (SQv * Dv)];
  h[i] = v;
  hbf[i] = (bf16)v;
}

// ---------------- GEMM: C[M,N] = A[M,K] * Wt[N,K]^T + bias ------------------
// A bf16 row-major, Wt bf16 [N][K]. TDM double-buffered staging; 8 waves;
// tile 128x128x32; wave tile 32x64 (2x4 WMMA).
__global__ __launch_bounds__(256) void k_gemm(const bf16* __restrict__ A, int lda,
                                              const bf16* __restrict__ Wt, int ldw,
                                              const float* __restrict__ bias,
                                              void* __restrict__ Cout, int ldc,
                                              bf16* __restrict__ vt, int vld, int vsplit,
                                              int K, int flags) {
  __shared__ __align__(32) bf16 As[2][128][32];
  __shared__ __align__(32) bf16 Bs[2][128][32];
  const int tid = threadIdx.x, lane = tid & 31, wave = tid >> 5;
  const int wm = wave & 3, wn = wave >> 2;
  const int hf = lane >> 4, rn = lane & 15;
  const int by = blockIdx.y, bx = blockIdx.x;

  v8f acc[2][4] = {};
  const int nk = K >> 5;
  const bf16* abase = A + (size_t)(by * 128) * lda;
  const bf16* wbase = Wt + (size_t)(bx * 128) * ldw;

  auto issue = [&](int kt, int buf) {
    if (wave == 0)
      tdm_load_2d(abase + kt * 32, (unsigned)(size_t)&As[buf][0][0],
                  (unsigned)lda, 128u, 32u, 128u, (unsigned long long)lda);
    else if (wave == 1)
      tdm_load_2d(wbase + kt * 32, (unsigned)(size_t)&Bs[buf][0][0],
                  (unsigned)ldw, 128u, 32u, 128u, (unsigned long long)ldw);
  };

  issue(0, 0);
  for (int kt = 0; kt < nk; ++kt) {
    const int cur = kt & 1;
    if (kt + 1 < nk) {
      issue(kt + 1, cur ^ 1);
      __builtin_amdgcn_s_wait_tensorcnt(1);   // current tile landed
    } else {
      __builtin_amdgcn_s_wait_tensorcnt(0);
    }
    __syncthreads();

    v16bf af[2], bw[4];
#pragma unroll
    for (int i = 0; i < 2; ++i)
      af[i] = *(const v16bf*)&As[cur][wm * 32 + i * 16 + rn][hf * 16];
#pragma unroll
    for (int j = 0; j < 4; ++j)
      bw[j] = *(const v16bf*)&Bs[cur][wn * 64 + j * 16 + rn][hf * 16];
#pragma unroll
    for (int i = 0; i < 2; ++i)
#pragma unroll
      for (int j = 0; j < 4; ++j)
        acc[i][j] = wmma_bf16(af[i], bw[j], acc[i][j]);
    __syncthreads();
  }

  // epilogue (C layout: lane rn -> col, vgpr r + 8*hf -> row)
#pragma unroll
  for (int i = 0; i < 2; ++i) {
#pragma unroll
    for (int j = 0; j < 4; ++j) {
      const int col = bx * 128 + wn * 64 + j * 16 + rn;
      const float bvv = bias[col];
#pragma unroll
      for (int r = 0; r < 8; ++r) {
        const int row = by * 128 + wm * 32 + i * 16 + r + 8 * hf;
        float v = acc[i][j][r] + bvv;
        if (flags & FLAG_GELU) v = gelu_f(v);
        if (flags & FLAG_KV_SPLIT) {
          if (col < vsplit)
            ((bf16*)Cout)[(size_t)row * ldc + col] = (bf16)v;          // K row-major
          else
            vt[(size_t)(col - vsplit) * vld + row] = (bf16)v;          // V col-major
        } else if (flags & FLAG_OUT_BF16) {
          ((bf16*)Cout)[(size_t)row * ldc + col] = (bf16)v;
        } else {
          ((float*)Cout)[(size_t)row * ldc + col] = v;
        }
      }
    }
  }
}

// ---------------- Flash attention: one wave per (b, h, 16-query tile) ------
// hq: [B*SQ][D] bf16 (raw h), kb: [B*SKV][D] bf16, vtb: [D][B*SKV] bf16.
__global__ __launch_bounds__(32) void k_attn(const bf16* __restrict__ hq,
                                             const bf16* __restrict__ kb,
                                             const bf16* __restrict__ vtb,
                                             bf16* __restrict__ outb) {
  __shared__ __align__(32) bf16 Qs[16][64];
  __shared__ __align__(32) bf16 Ks[2][32][64];
  __shared__ __align__(32) bf16 Vt[2][64][32];   // [hd][key]
  __shared__ __align__(32) bf16 Ps[16][32];

  const int t  = threadIdx.x;
  const int hf = t >> 4, rn = t & 15;
  int bid = blockIdx.x;
  const int b = bid / (Hv * QTv);
  bid -= b * Hv * QTv;
  const int hh = bid / QTv;
  const int q0 = (bid % QTv) * 16;

  // TDM: stage Q tile (16 rows x 64 hd)
  tdm_load_2d(hq + ((size_t)b * SQv + q0) * Dv + hh * HDv,
              (unsigned)(size_t)&Qs[0][0], Dv, 16u, 64u, 16u,
              (unsigned long long)Dv);

  auto issueKV = [&](int c, int buf) {
    const size_t kk0 = (size_t)c * 32;
    tdm_load_2d(kb + ((size_t)b * SKVv + kk0) * Dv + hh * HDv,
                (unsigned)(size_t)&Ks[buf][0][0], Dv, 32u, 64u, 32u,
                (unsigned long long)Dv);
    tdm_load_2d(vtb + (size_t)(hh * HDv) * NKV + (size_t)b * SKVv + kk0,
                (unsigned)(size_t)&Vt[buf][0][0], NKV, 64u, 32u, 64u,
                (unsigned long long)NKV);
  };
  issueKV(0, 0);

  __builtin_amdgcn_s_wait_tensorcnt(2);   // Q done (K0/V0 may be in flight)
  v16bf aq[2];
#pragma unroll
  for (int s = 0; s < 2; ++s) aq[s] = *(const v16bf*)&Qs[rn][s * 32 + hf * 16];

  float m_r[8], l_r[8];
#pragma unroll
  for (int r = 0; r < 8; ++r) { m_r[r] = -3.0e38f; l_r[r] = 0.f; }
  v8f o_acc[4] = {};

  const int NC = SKVv / 32;
  for (int c = 0; c < NC; ++c) {
    const int cur = c & 1;
    if (c + 1 < NC) {
      issueKV(c + 1, cur ^ 1);
      __builtin_amdgcn_s_wait_tensorcnt(2);   // chunk c landed
    } else {
      __builtin_amdgcn_s_wait_tensorcnt(0);
    }

    // S = Q K^T over HD=64 (two K=32 WMMAs) for two 16-key groups
    v8f sa = {}, sb = {};
#pragma unroll
    for (int s = 0; s < 2; ++s) {
      v16bf bk0 = *(const v16bf*)&Ks[cur][rn][s * 32 + hf * 16];
      sa = wmma_bf16(aq[s], bk0, sa);
      v16bf bk1 = *(const v16bf*)&Ks[cur][16 + rn][s * 32 + hf * 16];
      sb = wmma_bf16(aq[s], bk1, sb);
    }

    // online softmax (row = r + 8*hf, cols across lanes 0..15)
#pragma unroll
    for (int r = 0; r < 8; ++r) {
      float xa = sa[r] * 0.125f, xb = sb[r] * 0.125f;   // 1/sqrt(64)
      float mx = fmaxf(xa, xb);
#pragma unroll
      for (int o = 8; o > 0; o >>= 1) mx = fmaxf(mx, __shfl_xor(mx, o, 16));
      float mnew = fmaxf(m_r[r], mx);
      float fac = __expf(m_r[r] - mnew);
      float pa = __expf(xa - mnew), pb = __expf(xb - mnew);
      float sum = pa + pb;
#pragma unroll
      for (int o = 8; o > 0; o >>= 1) sum += __shfl_xor(sum, o, 16);
      l_r[r] = l_r[r] * fac + sum;
      m_r[r] = mnew;
#pragma unroll
      for (int j = 0; j < 4; ++j) o_acc[j][r] *= fac;
      Ps[r + 8 * hf][rn] = (bf16)pa;
      Ps[r + 8 * hf][16 + rn] = (bf16)pb;
    }
    __syncthreads();   // order Ps stores vs reloads

    // O += P V
    v16bf ap = *(const v16bf*)&Ps[rn][hf * 16];
#pragma unroll
    for (int j = 0; j < 4; ++j) {
      v16bf bv = *(const v16bf*)&Vt[cur][j * 16 + rn][hf * 16];
      o_acc[j] = wmma_bf16(ap, bv, o_acc[j]);
    }
  }

#pragma unroll
  for (int j = 0; j < 4; ++j)
#pragma unroll
    for (int r = 0; r < 8; ++r) {
      const int row = q0 + r + 8 * hf;
      const int col = hh * HDv + j * 16 + rn;
      outb[((size_t)b * SQv + row) * Dv + col] = (bf16)(o_acc[j][r] / l_r[r]);
    }
}

// ---------------- y = LayerNorm(x + r) * g + b  (+ optional bf16 shadow) ---
__global__ __launch_bounds__(256) void k_resid_ln(const float* __restrict__ x,
                                                  const float* __restrict__ r,
                                                  const float* __restrict__ g,
                                                  const float* __restrict__ bb,
                                                  float* __restrict__ y,
                                                  bf16* __restrict__ ybf) {
  __shared__ float red[8];
  const int row = blockIdx.x, tid = threadIdx.x;
  const float* xr = x + (size_t)row * Dv;
  const float* rr = r + (size_t)row * Dv;
  float v[4];
  float s = 0.f;
#pragma unroll
  for (int i = 0; i < 4; ++i) { v[i] = xr[tid + i * 256] + rr[tid + i * 256]; s += v[i]; }
#pragma unroll
  for (int o = 16; o > 0; o >>= 1) s += __shfl_xor(s, o, 32);
  if ((tid & 31) == 0) red[tid >> 5] = s;
  __syncthreads();
  float tot = 0.f;
#pragma unroll
  for (int i = 0; i < 8; ++i) tot += red[i];
  const float mean = tot * (1.0f / Dv);
  __syncthreads();
  float q = 0.f;
#pragma unroll
  for (int i = 0; i < 4; ++i) { float d = v[i] - mean; q += d * d; }
#pragma unroll
  for (int o = 16; o > 0; o >>= 1) q += __shfl_xor(q, o, 32);
  if ((tid & 31) == 0) red[tid >> 5] = q;
  __syncthreads();
  float qt = 0.f;
#pragma unroll
  for (int i = 0; i < 8; ++i) qt += red[i];
  const float inv = rsqrtf(qt * (1.0f / Dv) + 1e-5f);
  float* yr = y + (size_t)row * Dv;
#pragma unroll
  for (int i = 0; i < 4; ++i) {
    const int c = tid + i * 256;
    const float o = (v[i] - mean) * inv * g[c] + bb[c];
    yr[c] = o;
    if (ybf != nullptr) ybf[(size_t)row * Dv + c] = (bf16)o;
  }
}

// ---------------------------------------------------------------------------
extern "C" void kernel_launch(void* const* d_in, const int* in_sizes, int n_in,
                              void* d_out, int out_size, void* d_ws, size_t ws_size,
                              hipStream_t stream) {
  (void)in_sizes; (void)n_in; (void)out_size; (void)ws_size;
  const float* input_ids   = (const float*)d_in[1];
  const float* know        = (const float*)d_in[2];
  const float* pos_embed   = (const float*)d_in[3];
  const float* W_attn      = (const float*)d_in[4];
  const float* b_attn      = (const float*)d_in[5];
  const float* W_proj_attn = (const float*)d_in[6];
  const float* b_proj_attn = (const float*)d_in[7];
  const float* ln1_g       = (const float*)d_in[8];
  const float* ln1_b       = (const float*)d_in[9];
  const float* W_fc        = (const float*)d_in[10];
  const float* b_fc        = (const float*)d_in[11];
  const float* W_mlp       = (const float*)d_in[12];
  const float* b_mlp       = (const float*)d_in[13];
  const float* ln2_g       = (const float*)d_in[14];
  const float* ln2_b       = (const float*)d_in[15];

  char* ws = (char*)d_ws;
  size_t off = 0;
  auto alloc = [&](size_t bytes) -> void* {
    void* p = ws + off;
    off = (off + bytes + 255) & ~(size_t)255;
    return p;
  };

  // transposed bf16 weights [L][N][K]
  bf16* wkvT = (bf16*)alloc((size_t)Lv * 2 * Dv * Dv * sizeof(bf16));
  bf16* wpaT = (bf16*)alloc((size_t)Lv * Dv * Dv * sizeof(bf16));
  bf16* wfcT = (bf16*)alloc((size_t)Lv * 4 * Dv * Dv * sizeof(bf16));
  bf16* wmpT = (bf16*)alloc((size_t)Lv * Dv * 4 * Dv * sizeof(bf16));
  bf16* knb  = (bf16*)alloc((size_t)NKV * Dv * sizeof(bf16));       // know bf16
  bf16* kvb  = (bf16*)alloc((size_t)NKV * Dv * sizeof(bf16));       // K rows
  bf16* vtb  = (bf16*)alloc((size_t)Dv * NKV * sizeof(bf16));       // V^T
  float* hb  = (float*)alloc((size_t)Bv * SQv * Dv * sizeof(float));
  bf16*  hbh = (bf16*)alloc((size_t)Bv * SQv * Dv * sizeof(bf16));
  bf16*  ab  = (bf16*)alloc((size_t)Bv * SQv * Dv * sizeof(bf16));  // attn out
  float* pb  = (float*)alloc((size_t)Bv * SQv * Dv * sizeof(float));
  bf16*  pbh = (bf16*)alloc((size_t)Bv * SQv * Dv * sizeof(bf16));
  float* mb  = (float*)alloc((size_t)Bv * SQv * Dv * sizeof(float));
  float* tb  = (float*)alloc((size_t)Bv * SQv * Dv * sizeof(float)); // proj out
  bf16*  tbh = (bf16*)alloc((size_t)Bv * SQv * 4 * Dv * sizeof(bf16)); // fc out

  auto cvtT = [&](const float* s, bf16* d, int K, int N, int stride, int coff) {
    long total = (long)Lv * N * K;
    k_cvtT<<<dim3((unsigned)((total + 255) / 256)), dim3(256), 0, stream>>>(
        s, d, total, K, N, stride, coff);
  };
  cvtT(W_attn, wkvT, Dv, 2 * Dv, 3 * Dv, Dv);     // skip discarded Q third
  cvtT(W_proj_attn, wpaT, Dv, Dv, Dv, 0);
  cvtT(W_fc, wfcT, Dv, 4 * Dv, 4 * Dv, 0);
  cvtT(W_mlp, wmpT, 4 * Dv, Dv, Dv, 0);
  {
    long total = (long)NKV * Dv;
    k_cvt<<<dim3((unsigned)((total + 255) / 256)), dim3(256), 0, stream>>>(know, knb, total);
  }
  {
    int total = Bv * SQv * Dv;
    k_add_pos<<<dim3((total + 255) / 256), dim3(256), 0, stream>>>(
        input_ids, pos_embed, hb, hbh, total);
  }

  for (int l = 0; l < Lv; ++l) {
    // 1. [k|v] = know @ W_attn[:, D:3D] + b : K rows to kvb, V col-major to vtb
    k_gemm<<<dim3(2 * Dv / 128, NKV / 128), dim3(256), 0, stream>>>(
        knb, Dv, wkvT + (size_t)l * 2 * Dv * Dv, Dv,
        b_attn + (size_t)l * 3 * Dv + Dv, kvb, Dv,
        vtb, NKV, Dv, Dv, FLAG_KV_SPLIT);
    // 2. attention (q = raw h, bf16)
    k_attn<<<dim3(Bv * Hv * QTv), dim3(32), 0, stream>>>(hbh, kvb, vtb, ab);
    // 3. attention output projection -> tb (fp32)
    k_gemm<<<dim3(Dv / 128, Bv * SQv / 128), dim3(256), 0, stream>>>(
        ab, Dv, wpaT + (size_t)l * Dv * Dv, Dv,
        b_proj_attn + (size_t)l * Dv, tb, Dv,
        nullptr, 0, 0, Dv, 0);
    // 4. p = LN1(h + attn)  (fp32 + bf16 shadow)
    k_resid_ln<<<dim3(Bv * SQv), dim3(256), 0, stream>>>(
        hb, tb, ln1_g + (size_t)l * Dv, ln1_b + (size_t)l * Dv, pb, pbh);
    // 5. fc + gelu -> tbh (bf16)
    k_gemm<<<dim3(4 * Dv / 128, Bv * SQv / 128), dim3(256), 0, stream>>>(
        pbh, Dv, wfcT + (size_t)l * 4 * Dv * Dv, Dv,
        b_fc + (size_t)l * 4 * Dv, tbh, 4 * Dv,
        nullptr, 0, 0, Dv, FLAG_GELU | FLAG_OUT_BF16);
    // 6. mlp projection -> mb (fp32)
    k_gemm<<<dim3(Dv / 128, Bv * SQv / 128), dim3(256), 0, stream>>>(
        tbh, 4 * Dv, wmpT + (size_t)l * Dv * 4 * Dv, 4 * Dv,
        b_mlp + (size_t)l * Dv, mb, Dv,
        nullptr, 0, 0, 4 * Dv, 0);
    // 7. h = LN2(p + m)  (last layer -> d_out)
    float* outp = (l == Lv - 1) ? (float*)d_out : hb;
    bf16*  outbf = (l == Lv - 1) ? nullptr : hbh;
    k_resid_ln<<<dim3(Bv * SQv), dim3(256), 0, stream>>>(
        pb, mb, ln2_g + (size_t)l * Dv, ln2_b + (size_t)l * Dv, outp, outbf);
  }
}
